// LinearAppearance_68092411510798
// MI455X (gfx1250) — matrix operand order repaired and checked by reading, not compiled, not run on hardware
//
#include <hip/hip_runtime.h>
#include <hip/hip_bf16.h>
#include <type_traits>

// Problem constants (from reference)
#define B_    8
#define N_    256
#define T_    50
#define FEAT_ 1799
#define TOK_  128

#define KC       32                       // K per bf16 WMMA
#define NCHUNK   ((FEAT_ + KC - 1) / KC)  // 57
#define TAIL_K0  (FEAT_ - KC)             // 1767: shifted in-bounds tail window
#define KVALID   ((NCHUNK - 1) * KC - TAIL_K0)  // 25: window pos >= 25 is new data
#define A_STRIDE 40                       // bf16 elems per A row in LDS (32 + pad)
#define B_STRIDE 40                       // bf16 elems per Bt row in LDS
#define TOK_STRIDE 132                    // floats per tokens row (128 + pad)

typedef __attribute__((ext_vector_type(16))) __bf16 v16bf;
typedef __attribute__((ext_vector_type(8)))  float  v8f;

union Frag {
  uint4 q[2];
  v16bf v;
};

__global__ __launch_bounds__(256) void linear_ema_wmma_kernel(
    const float* __restrict__ feats,          // [B,N,T,FEAT] f32
    const unsigned char* __restrict__ masks,  // [B,N,T] bool (1 byte)
    const float* __restrict__ W,              // [TOK,FEAT] f32
    const float* __restrict__ bias,           // [TOK] f32
    float* __restrict__ out)                  // [B,N,TOK] f32
{
  __shared__ alignas(16) __bf16 Alds[64 * A_STRIDE];    //  5 KB
  __shared__ alignas(16) __bf16 Blds[TOK_ * B_STRIDE];  // 10 KB
  __shared__ float tokLds[64 * TOK_STRIDE];             // 33 KB
  __shared__ float maskF[T_];

  const int tid = threadIdx.x;
  const int bn  = blockIdx.x;                          // 0 .. B*N-1
  const float* feats_bn = feats + (size_t)bn * T_ * FEAT_;

  // Preload mask row once as float multipliers (covered by loop barriers).
  if (tid < T_) maskF[tid] = masks[(size_t)bn * T_ + tid] ? 1.0f : 0.0f;

  const int lane   = tid & 31;                         // wave32
  const int waveId = tid >> 5;                         // 0..7
  const int mW  = (waveId & 3) * 16;                   // 4 waves along M (t)
  const int nW  = (waveId >> 2) * 64;                  // 2 waves along N (tok)
  const int n16 = lane & 15;
  const int g   = lane >> 4;                           // lane half

  v8f acc[4] = {};                                     // 16x64 output tile per wave

  // A staging: 64 rows x 32 k, 8 elems/thread. Rows >= T: load clamped row 0
  // (always in-bounds, finite) and scale by 0.0 -> branch-free zero padding.
  const int am    = tid >> 2;
  const int asub  = tid & 3;
  const float rowScale = (am < T_) ? 1.0f : 0.0f;
  const float* aBase = feats_bn + (size_t)(am < T_ ? am : 0) * FEAT_;
  __bf16* ad = Alds + am * A_STRIDE + asub * 8;
  // B staging: 128 rows (tok) x 32 k, 16 elems/thread
  const int wn    = tid >> 1;
  const int wsub  = tid & 1;
  const float* bBase = W + (size_t)wn * FEAT_;
  __bf16* bd = Blds + wn * B_STRIDE + wsub * 16;

  // Fragment lane pointers (ISA 7.12.2 16-bit A layout; B row-striped across lanes)
  const uint4* paF = reinterpret_cast<const uint4*>(Alds + (mW + n16) * A_STRIDE + 8 * g);
  const uint4* pbF[4];
  #pragma unroll
  for (int j = 0; j < 4; ++j)
    pbF[j] = reinterpret_cast<const uint4*>(Blds + (nW + j * 16 + n16) * B_STRIDE + 8 * g);

  auto chunk = [&](int k0, auto tailTag) {
    constexpr bool TAIL = decltype(tailTag)::value;

    // ---- global -> registers. ALL loads unconditional & contiguous (b128-able).
    // Tail uses a shifted, fully in-bounds window [FEAT-32, FEAT); window
    // positions w < KVALID were already accumulated -> scaled to exact 0.0.
    float av[8];
    #pragma unroll
    for (int i = 0; i < 8; ++i) {
      const int w = asub * 8 + i;
      float s = rowScale;
      if constexpr (TAIL) s = (w >= KVALID) ? rowScale : 0.0f;   // v_cndmask
      av[i] = aBase[k0 + w] * s;
    }
    float bv[16];
    #pragma unroll
    for (int i = 0; i < 16; ++i) {
      const int w = wsub * 16 + i;
      const float x = bBase[k0 + w];
      if constexpr (TAIL) bv[i] = (w >= KVALID) ? x : 0.0f;
      else                bv[i] = x;
    }
    if constexpr (!TAIL) {                             // global_prefetch_b8 (speculative OK)
      __builtin_prefetch(aBase + k0 + asub * 8 + KC, 0, 3);
      __builtin_prefetch(bBase + k0 + wsub * 16 + KC, 0, 3);
    }

    // ---- registers -> LDS as bf16 (branch-free; backend packs into b128) ----
    __syncthreads();                                   // previous iteration's frag reads done
    #pragma unroll
    for (int i = 0; i < 8; ++i)
      ad[i] = (__bf16)av[i];
    #pragma unroll
    for (int i = 0; i < 16; ++i)
      bd[i] = (__bf16)bv[i];
    __syncthreads();

    // ---- fragment loads (batched), then 4 WMMAs back-to-back ----
    Frag fa;
    fa.q[0] = paF[0];
    fa.q[1] = paF[2];
    Frag fb[4];
    #pragma unroll
    for (int j = 0; j < 4; ++j) {
      fb[j].q[0] = pbF[j][0];
      fb[j].q[1] = pbF[j][2];
    }
    #pragma unroll
    for (int j = 0; j < 4; ++j)
      acc[j] = __builtin_amdgcn_wmma_f32_16x16x32_bf16(
          false, fa.v, false, fb[j].v, (short)0, acc[j], false, false);
  };

  for (int c = 0; c < NCHUNK - 1; ++c) chunk(c * KC, std::false_type{});
  chunk(TAIL_K0, std::true_type{});                    // shifted in-bounds K tail

  // ---- accumulators -> LDS tokens[64][128] ----
  __syncthreads();
  #pragma unroll
  for (int j = 0; j < 4; ++j) {
    #pragma unroll
    for (int r = 0; r < 8; ++r) {
      // D layout: lane half g -> M rows r + 8g, lane n16 -> column
      tokLds[(mW + r + 8 * g) * TOK_STRIDE + nW + j * 16 + n16] = acc[j][r];
    }
  }
  __syncthreads();

  // ---- branchless masked EMA fold over t = T-1 .. 0 (reference scan order) ----
  if (tid < TOK_) {
    const float bsc = bias[tid];
    float ema = 0.0f;
    for (int t = T_ - 1; t >= 0; --t) {
      const float tok = tokLds[t * TOK_STRIDE + tid] + bsc;
      const float mm  = maskF[t];
      // m=1: 0.9*tok + 0.1*ema ; m=0: ema  (== m*(a*tok+(1-a)*e)+(1-m)*e)
      ema = fmaf(mm * 0.9f, tok - ema, ema);
    }
    out[(size_t)bn * TOK_ + tid] = ema;
  }
}

extern "C" void kernel_launch(void* const* d_in, const int* in_sizes, int n_in,
                              void* d_out, int out_size, void* d_ws, size_t ws_size,
                              hipStream_t stream) {
  (void)in_sizes; (void)n_in; (void)out_size; (void)d_ws; (void)ws_size;
  const float*         feats = (const float*)d_in[0];
  const unsigned char* masks = (const unsigned char*)d_in[1];   // jax bool -> 1 byte
  const float*         W     = (const float*)d_in[2];
  const float*         bias  = (const float*)d_in[3];
  float*               out   = (float*)d_out;

  dim3 grid(B_ * N_);   // 2048 workgroups, one per (b,n)
  dim3 block(256);      // 8 waves (wave32)
  linear_ema_wmma_kernel<<<grid, block, 0, stream>>>(feats, masks, W, bias, out);
}
